// BinaryCNN_78580721647854
// MI455X (gfx1250) — 8-bit matrix operand order repaired and checked by reading, compiled, not run on hardware
//
#include <hip/hip_runtime.h>
#include <stdint.h>

typedef __attribute__((ext_vector_type(8))) int v8i;

#define DEV __device__ __forceinline__

#define BATCH 2048
#define PIX   784
#define IMH   28
#define IMW   28
#define C1    32
#define C2    64
#define PH    13
#define PPIX  169
#define K2    288
#define K2C   5
#define UNITS 2048
#define KFC   10816
#define KFCC  169
#define KC4   32
#define OUTC  10
#define EPSN  1e-4f

DEV float ap2f(float x) {
    float ax = fabsf(x) + 1e-12f;
    float e  = rintf(log2f(ax));
    return copysignf(exp2f(e), x);
}
DEV float sgnf(float x) { return (x > 0.f) ? 1.f : ((x < 0.f) ? -1.f : 0.f); }
DEV int8_t sgnb(float x) { return (x > 0.f) ? (int8_t)1 : ((x < 0.f) ? (int8_t)-1 : (int8_t)0); }

DEV void a_swz(int m, int ko, int& lane, int& v, int& byt) {
    int g = ko >> 5, hf = (ko >> 4) & 1, w = ko & 15;
    v = g * 4 + (w >> 2);
    byt = w & 3;
    lane = hf * 16 + m;
}

__global__ void k_zero(float* p, int n) {
    int i = blockIdx.x * blockDim.x + threadIdx.x;
    if (i < n) p[i] = 0.f;
}

__global__ void k_conv1(const float* __restrict__ x, const float* __restrict__ w1,
                        const float* __restrict__ b1, float* __restrict__ h1) {
    size_t id = (size_t)blockIdx.x * blockDim.x + threadIdx.x;
    if (id >= (size_t)BATCH * PIX * C1) return;
    int c = (int)(id & 31);
    size_t pix = id >> 5;
    int b = (int)(pix / PIX), rem = (int)(pix % PIX);
    int y = rem / IMW, xq = rem % IMW;
    float acc = b1[c];
#pragma unroll
    for (int t = 0; t < 9; ++t) {
        int iy = y + t / 3 - 1, ix = xq + t % 3 - 1;
        if (iy >= 0 && iy < IMH && ix >= 0 && ix < IMW)
            acc += sgnf(w1[c * 9 + t]) * x[(size_t)b * PIX + iy * IMW + ix];
    }
    h1[pix * C1 + c] = fmaxf(acc, 0.f);
}

template <int C>
__global__ void k_chan_sum(const float* __restrict__ x, size_t n, float* __restrict__ gsum) {
    __shared__ float sm[256];
    size_t stride = (size_t)gridDim.x * blockDim.x;
    size_t g = (size_t)blockIdx.x * blockDim.x + threadIdx.x;
    float acc = 0.f;
    for (size_t i = g; i < n; i += stride) acc += x[i];
    sm[threadIdx.x] = acc;
    __syncthreads();
    if (threadIdx.x < C) {
        float s = 0.f;
        for (int j = threadIdx.x; j < 256; j += C) s += sm[j];
        unsafeAtomicAdd(&gsum[threadIdx.x], s);
    }
}

template <int C>
__global__ void k_chan_var(const float* __restrict__ x, size_t n, const float* __restrict__ gsum,
                           float cnt, float* __restrict__ gvs) {
    __shared__ float sm[256];
    size_t stride = (size_t)gridDim.x * blockDim.x;
    size_t g = (size_t)blockIdx.x * blockDim.x + threadIdx.x;
    float mean = gsum[threadIdx.x % C] / cnt;
    float acc = 0.f;
    for (size_t i = g; i < n; i += stride) { float d = x[i] - mean; acc += d * ap2f(d); }
    sm[threadIdx.x] = acc;
    __syncthreads();
    if (threadIdx.x < C) {
        float s = 0.f;
        for (int j = threadIdx.x; j < 256; j += C) s += sm[j];
        unsafeAtomicAdd(&gvs[threadIdx.x], s);
    }
}

__global__ void k_bin1(const float* __restrict__ h1, const float* __restrict__ gsum,
                       const float* __restrict__ gvs, const float* __restrict__ ga,
                       const float* __restrict__ be, float cnt, int8_t* __restrict__ out) {
    size_t id = (size_t)blockIdx.x * blockDim.x + threadIdx.x;
    if (id >= (size_t)BATCH * PIX * C1) return;
    int c = (int)(id & 31);
    float mean = gsum[c] / cnt, var = gvs[c] / cnt;
    float v = h1[id] - mean;
    float o = ap2f(ga[c]) * (v * ap2f(1.f / sqrtf(var + EPSN))) + be[c];
    out[id] = sgnb(o);
}

__global__ void k_prep_bf2(const float* __restrict__ w2, uint32_t* __restrict__ bf) {
    int idx = blockIdx.x * blockDim.x + threadIdx.x;
    if (idx >= 4 * K2C * 32 * 8) return;
    int v = idx & 7, lane = (idx >> 3) & 31;
    int rest = idx >> 8, kc = rest % K2C, nt = rest / K2C;
    int hf = lane >> 4, n = nt * 16 + (lane & 15);
    int kbase = kc * 64 + (v >> 2) * 32 + hf * 16 + (v & 3) * 4;
    uint32_t dw = 0;
    for (int j = 0; j < 4; ++j) {
        int K = kbase + j;
        int8_t sb = 0;
        if (K < K2) {
            int tap = K >> 5, c = K & 31;
            sb = sgnb(w2[(((size_t)n * C1 + c) * 3 + tap / 3) * 3 + tap % 3]);
        }
        dw |= ((uint32_t)(uint8_t)sb) << (8 * j);
    }
    bf[idx] = dw;
}

__global__ void __launch_bounds__(128) k_conv2(const int8_t* __restrict__ h1b,
                                               const v8i* __restrict__ bf2,
                                               const float* __restrict__ b2,
                                               float* __restrict__ out) {
    __shared__ uint32_t sm[64 * 80];
    int tid = threadIdx.x;
    size_t pixbase = (size_t)blockIdx.x * 64;
    for (int i = tid; i < 64 * 80; i += 128) {
        int row = i / 80, d = i % 80, Kb = d * 4;
        uint32_t val = 0;
        if (Kb < K2) {
            int pix = (int)(pixbase + row);
            int b = pix / PIX, rem = pix % PIX, y = rem / IMW, xq = rem % IMW;
            int tap = Kb >> 5, c0 = Kb & 31;
            int iy = y + tap / 3 - 1, ix = xq + tap % 3 - 1;
            if (iy >= 0 && iy < IMH && ix >= 0 && ix < IMW)
                val = *(const uint32_t*)(h1b + (((size_t)b * PIX + iy * IMW + ix) * C1 + c0));
        }
        sm[i] = val;
    }
    __syncthreads();
    int wave = tid >> 5, lane = tid & 31;
    int hf = lane >> 4, m = lane & 15, nn = lane & 15;
    v8i acc[4] = {};
    for (int kc = 0; kc < K2C; ++kc) {
        v8i a;
#pragma unroll
        for (int v = 0; v < 8; ++v) {
            int Kb = kc * 64 + (v >> 2) * 32 + hf * 16 + (v & 3) * 4;
            a[v] = (int)sm[(wave * 16 + m) * 80 + (Kb >> 2)];
        }
#pragma unroll
        for (int nt = 0; nt < 4; ++nt) {
            v8i bfr = bf2[(size_t)(nt * K2C + kc) * 32 + lane];
            acc[nt] = __builtin_amdgcn_wmma_i32_16x16x64_iu8(true, a, true, bfr, acc[nt],
                                                             false, false);
        }
    }
#pragma unroll
    for (int nt = 0; nt < 4; ++nt)
#pragma unroll
        for (int r = 0; r < 8; ++r) {
            size_t pix = pixbase + wave * 16 + hf * 8 + r;
            int col = nt * 16 + nn;
            out[pix * C2 + col] = fmaxf((float)acc[nt][r] + b2[col], 0.f);
        }
}

__global__ void k_pool(const float* __restrict__ in, float* __restrict__ out) {
    size_t id = (size_t)blockIdx.x * blockDim.x + threadIdx.x;
    if (id >= (size_t)BATCH * PPIX * C2) return;
    int c = (int)(id & 63);
    size_t rem = id >> 6;
    int xo = (int)(rem % PH); rem /= PH;
    int yo = (int)(rem % PH);
    int b = (int)(rem / PH);
    float mx = -3.402823466e38f;
#pragma unroll
    for (int dy = 0; dy < 3; ++dy)
#pragma unroll
        for (int dx = 0; dx < 3; ++dx) {
            int iy = 2 * yo + dy, ix = 2 * xo + dx;
            mx = fmaxf(mx, in[((size_t)b * PIX + iy * IMW + ix) * C2 + c]);
        }
    out[id] = mx;
}

__global__ void k_bin2(const float* __restrict__ pooled, const float* __restrict__ gsum,
                       const float* __restrict__ gvs, const float* __restrict__ ga,
                       const float* __restrict__ be, float cnt, int8_t* __restrict__ frag) {
    size_t id = (size_t)blockIdx.x * blockDim.x + threadIdx.x;
    if (id >= (size_t)BATCH * PPIX * C2) return;
    int c = (int)(id & 63);
    size_t rem = id >> 6;
    int xo = (int)(rem % PH); rem /= PH;
    int yo = (int)(rem % PH);
    int b = (int)(rem / PH);
    float mean = gsum[c] / cnt, var = gvs[c] / cnt;
    float v = pooled[id] - mean;
    float o = ap2f(ga[c]) * (v * ap2f(1.f / sqrtf(var + EPSN))) + be[c];
    int8_t s = sgnb(o);
    int k = c * PPIX + yo * PH + xo;
    int kc = k >> 6, ko = k & 63;
    int lane, vv, byt;
    a_swz(b & 15, ko, lane, vv, byt);
    size_t addr = ((size_t)((b >> 4) * KFCC + kc) * 32 + lane) * 32 + vv * 4 + byt;
    frag[addr] = s;
}

__global__ void k_prep_bf3(const float* __restrict__ w3, uint32_t* __restrict__ bf) {
    int idx = blockIdx.x * blockDim.x + threadIdx.x;
    if (idx >= 128 * KFCC * 32 * 8) return;
    int v = idx & 7, lane = (idx >> 3) & 31;
    int rest = idx >> 8, kc = rest % KFCC, nt = rest / KFCC;
    int hf = lane >> 4, n = nt * 16 + (lane & 15);
    int kbase = kc * 64 + (v >> 2) * 32 + hf * 16 + (v & 3) * 4;
    uint32_t dw = 0;
    for (int j = 0; j < 4; ++j) {
        int8_t sb = sgnb(w3[(size_t)n * KFC + kbase + j]);
        dw |= ((uint32_t)(uint8_t)sb) << (8 * j);
    }
    bf[idx] = dw;
}

__global__ void __launch_bounds__(128) k_fc3(const v8i* __restrict__ af,
                                             const v8i* __restrict__ bf,
                                             const float* __restrict__ b3,
                                             float* __restrict__ h3) {
    int wave = threadIdx.x >> 5, lane = threadIdx.x & 31;
    int mtile = blockIdx.x * 4 + wave;
    int ntb = blockIdx.y * 4;
    v8i acc[4] = {};
    for (int kc = 0; kc < KFCC; ++kc) {
        v8i a = af[((size_t)mtile * KFCC + kc) * 32 + lane];
#pragma unroll
        for (int nt = 0; nt < 4; ++nt) {
            v8i b = bf[((size_t)(ntb + nt) * KFCC + kc) * 32 + lane];
            acc[nt] = __builtin_amdgcn_wmma_i32_16x16x64_iu8(true, a, true, b, acc[nt],
                                                             false, false);
        }
    }
    int hf = lane >> 4, nn = lane & 15;
#pragma unroll
    for (int nt = 0; nt < 4; ++nt)
#pragma unroll
        for (int r = 0; r < 8; ++r) {
            int row = mtile * 16 + hf * 8 + r;
            int col = (ntb + nt) * 16 + nn;
            h3[(size_t)row * UNITS + col] = fmaxf((float)acc[nt][r] + b3[col], 0.f);
        }
}

__global__ void k_col_sum(const float* __restrict__ x, float* __restrict__ out) {
    int u = blockIdx.x * blockDim.x + threadIdx.x;
    if (u >= UNITS) return;
    float acc = 0.f;
    for (int r = 0; r < BATCH; ++r) acc += x[(size_t)r * UNITS + u];
    out[u] = acc;
}
__global__ void k_col_var(const float* __restrict__ x, const float* __restrict__ gsum,
                          float* __restrict__ out) {
    int u = blockIdx.x * blockDim.x + threadIdx.x;
    if (u >= UNITS) return;
    float mean = gsum[u] / (float)BATCH;
    float acc = 0.f;
    for (int r = 0; r < BATCH; ++r) { float d = x[(size_t)r * UNITS + u] - mean; acc += d * ap2f(d); }
    out[u] = acc;
}

__global__ void k_bin3(const float* __restrict__ h3, const float* __restrict__ gsum,
                       const float* __restrict__ gvs, const float* __restrict__ ga,
                       const float* __restrict__ be, int8_t* __restrict__ frag) {
    size_t id = (size_t)blockIdx.x * blockDim.x + threadIdx.x;
    if (id >= (size_t)BATCH * UNITS) return;
    int u = (int)(id & (UNITS - 1));
    int b = (int)(id >> 11);
    float mean = gsum[u] / (float)BATCH, var = gvs[u] / (float)BATCH;
    float v = h3[id] - mean;
    float o = ap2f(ga[u]) * (v * ap2f(1.f / sqrtf(var + EPSN))) + be[u];
    int8_t s = sgnb(o);
    int kc = u >> 6, ko = u & 63;
    int lane, vv, byt;
    a_swz(b & 15, ko, lane, vv, byt);
    size_t addr = ((size_t)((b >> 4) * KC4 + kc) * 32 + lane) * 32 + vv * 4 + byt;
    frag[addr] = s;
}

__global__ void k_prep_bf4(const float* __restrict__ w4, uint32_t* __restrict__ bf) {
    int idx = blockIdx.x * blockDim.x + threadIdx.x;
    if (idx >= KC4 * 32 * 8) return;
    int v = idx & 7, lane = (idx >> 3) & 31;
    int kc = idx >> 8;
    int hf = lane >> 4, n = lane & 15;
    int kbase = kc * 64 + (v >> 2) * 32 + hf * 16 + (v & 3) * 4;
    uint32_t dw = 0;
    if (n < OUTC)
        for (int j = 0; j < 4; ++j) {
            int8_t sb = sgnb(w4[(size_t)n * UNITS + kbase + j]);
            dw |= ((uint32_t)(uint8_t)sb) << (8 * j);
        }
    bf[idx] = dw;
}

__global__ void __launch_bounds__(128) k_fc4(const v8i* __restrict__ af,
                                             const v8i* __restrict__ bf,
                                             const float* __restrict__ b4,
                                             float* __restrict__ out) {
    int wave = threadIdx.x >> 5, lane = threadIdx.x & 31;
    int mtile = blockIdx.x * 4 + wave;
    v8i acc = {};
    for (int kc = 0; kc < KC4; ++kc) {
        v8i a = af[((size_t)mtile * KC4 + kc) * 32 + lane];
        v8i b = bf[(size_t)kc * 32 + lane];
        acc = __builtin_amdgcn_wmma_i32_16x16x64_iu8(true, a, true, b, acc, false, false);
    }
    int hf = lane >> 4, nn = lane & 15;
    if (nn < OUTC) {
#pragma unroll
        for (int r = 0; r < 8; ++r) {
            int row = mtile * 16 + hf * 8 + r;
            out[(size_t)row * OUTC + nn] = (float)acc[r] + b4[nn];
        }
    }
}

extern "C" void kernel_launch(void* const* d_in, const int* in_sizes, int n_in,
                              void* d_out, int out_size, void* d_ws, size_t ws_size,
                              hipStream_t stream) {
    const float* x   = (const float*)d_in[0];
    const float* w1  = (const float*)d_in[1];
    const float* b1  = (const float*)d_in[2];
    const float* ga1 = (const float*)d_in[3];
    const float* bt1 = (const float*)d_in[4];
    const float* w2  = (const float*)d_in[5];
    const float* b2  = (const float*)d_in[6];
    const float* ga2 = (const float*)d_in[7];
    const float* bt2 = (const float*)d_in[8];
    const float* w3  = (const float*)d_in[9];
    const float* b3  = (const float*)d_in[10];
    const float* ga3 = (const float*)d_in[11];
    const float* bt3 = (const float*)d_in[12];
    const float* w4  = (const float*)d_in[13];
    const float* b4  = (const float*)d_in[14];

    char* ws = (char*)d_ws;
    float*    h1     = (float*)(ws + 0);
    float*    c2out  = (float*)(ws + 0);
    int8_t*   h1b    = (int8_t*)(ws + 411041792ULL);
    float*    pooled = (float*)(ws + 462422016ULL);
    int8_t*   h2bf   = (int8_t*)(ws + 551026688ULL);
    int8_t*   bf3    = (int8_t*)(ws + 573177856ULL);
    float*    h3     = (float*)(ws + 595329024ULL);
    int8_t*   h3bf   = (int8_t*)(ws + 612106240ULL);
    uint32_t* bf2    = (uint32_t*)(ws + 616300544ULL);
    uint32_t* bf4    = (uint32_t*)(ws + 616321024ULL);
    float*    stats  = (float*)(ws + 616353792ULL);
    float* sum1 = stats;         float* vs1 = stats + 32;
    float* sum2 = stats + 64;    float* vs2 = stats + 128;
    float* sum3 = stats + 192;   float* vs3 = stats + 192 + 2048;

    size_t n1 = (size_t)BATCH * PIX * C1;
    size_t n2 = (size_t)BATCH * PPIX * C2;

    k_zero<<<17, 256, 0, stream>>>(stats, 4288);
    k_conv1<<<(unsigned)(n1 / 256), 256, 0, stream>>>(x, w1, b1, h1);
    k_chan_sum<C1><<<2048, 256, 0, stream>>>(h1, n1, sum1);
    k_chan_var<C1><<<2048, 256, 0, stream>>>(h1, n1, sum1, (float)(BATCH * PIX), vs1);
    k_bin1<<<(unsigned)(n1 / 256), 256, 0, stream>>>(h1, sum1, vs1, ga1, bt1,
                                                     (float)(BATCH * PIX), h1b);
    k_prep_bf2<<<20, 256, 0, stream>>>(w2, bf2);
    k_conv2<<<(BATCH * PIX) / 64, 128, 0, stream>>>(h1b, (const v8i*)bf2, b2, c2out);
    k_pool<<<(unsigned)(n2 / 256), 256, 0, stream>>>(c2out, pooled);
    k_chan_sum<C2><<<2048, 256, 0, stream>>>(pooled, n2, sum2);
    k_chan_var<C2><<<2048, 256, 0, stream>>>(pooled, n2, sum2, (float)(BATCH * PPIX), vs2);
    k_bin2<<<(unsigned)(n2 / 256), 256, 0, stream>>>(pooled, sum2, vs2, ga2, bt2,
                                                     (float)(BATCH * PPIX), h2bf);
    k_prep_bf3<<<(128 * KFCC * 32 * 8) / 256, 256, 0, stream>>>(w3, (uint32_t*)bf3);
    dim3 grid_fc3(32, 32);
    k_fc3<<<grid_fc3, 128, 0, stream>>>((const v8i*)h2bf, (const v8i*)bf3, b3, h3);
    k_col_sum<<<8, 256, 0, stream>>>(h3, sum3);
    k_col_var<<<8, 256, 0, stream>>>(h3, sum3, vs3);
    k_bin3<<<(unsigned)(((size_t)BATCH * UNITS) / 256), 256, 0, stream>>>(h3, sum3, vs3, ga3,
                                                                          bt3, h3bf);
    k_prep_bf4<<<32, 256, 0, stream>>>(w4, bf4);
    k_fc4<<<32, 128, 0, stream>>>((const v8i*)h3bf, (const v8i*)bf4, b4, (float*)d_out);
}
